// SelfAttentionBlock_4260607557728
// MI455X (gfx1250) — compile-verified, matrix operand order checked by reading, not run on hardware
//
#include <hip/hip_runtime.h>
#include <hip/hip_bf16.h>
#include <hip/hip_fp16.h>

#define HEADS    16
#define HEAD_DIM 64
#define EMBED    1024
#define NB       4
#define SEQ      2048
#define KT       64      // keys per attention tile
#define NWAVE    4       // waves per attention block

typedef __attribute__((ext_vector_type(16))) _Float16 v16h;
typedef __attribute__((ext_vector_type(8)))  _Float16 v8h;
typedef __attribute__((ext_vector_type(8)))  float    v8f;
typedef __attribute__((ext_vector_type(4)))  float    v4f;
typedef unsigned int u32x4 __attribute__((ext_vector_type(4)));
typedef int          i32x4 __attribute__((ext_vector_type(4)));
typedef int          i32x8 __attribute__((ext_vector_type(8)));

// D = A(16x32 f16) * B(32x16 f16) + C(16x16 f32)
static __device__ __forceinline__ v8f wmma_f16(v16h a, v16h b, v8f c) {
  return __builtin_amdgcn_wmma_f32_16x16x32_f16(false, a, false, b, (short)0, c, false, false);
}

// A-fragment (16x32) from a row-major f16 matrix; also serves as B-fragment
// when given B^T row-major (lane&15 selects the B column).
// ISA layout: lane L (m = L&15, hi = L>>4): elems 0..7 = K = 8*hi + 0..7,
// elems 8..15 = K = 16 + 8*hi + 0..7  -> two contiguous 16-byte loads.
static __device__ __forceinline__ v16h frag_f16(const _Float16* p,
                                                int ld, int row0, int col0) {
  const int lane = threadIdx.x & 31;
  const int hi8  = (lane >> 4) << 3;
  const _Float16* base = p + (size_t)(row0 + (lane & 15)) * ld + col0;
  v8h lo = *(const v8h*)(base + hi8);
  v8h hh = *(const v8h*)(base + 16 + hi8);
  v16h out;
#pragma unroll
  for (int i = 0; i < 8; ++i) { out[i] = lo[i]; out[8 + i] = hh[i]; }
  return out;
}

// Same fragment sourced from f32 memory, converted to f16 in registers.
static __device__ __forceinline__ v16h frag_f32(const float* __restrict__ p,
                                                int ld, int row0, int col0) {
  const int lane = threadIdx.x & 31;
  const int hi8  = (lane >> 4) << 3;
  const float* base = p + (size_t)(row0 + (lane & 15)) * ld + col0;
  v4f a0 = *(const v4f*)(base + hi8);
  v4f a1 = *(const v4f*)(base + hi8 + 4);
  v4f b0 = *(const v4f*)(base + 16 + hi8);
  v4f b1 = *(const v4f*)(base + 16 + hi8 + 4);
  v16h out;
#pragma unroll
  for (int i = 0; i < 4; ++i) {
    out[i]      = (_Float16)a0[i];
    out[4 + i]  = (_Float16)a1[i];
    out[8 + i]  = (_Float16)b0[i];
    out[12 + i] = (_Float16)b1[i];
  }
  return out;
}

// ---- Tensor Data Mover: DMA a (tile_d0 x tile_d1) f16 tile (row-major,
// row stride stride0 elems) from global memory into LDS at lds_off. ----------
static __device__ __forceinline__ unsigned lds_byte_off(const void* p) {
  // generic pointers to LDS carry the LDS byte address in their low 32 bits
  return (unsigned)(unsigned long long)(uintptr_t)p;
}

static __device__ __forceinline__ void tdm_load_f16(
    unsigned lds_off, const void* gptr,
    unsigned tile_d0, unsigned tile_d1,
    unsigned tensor_d0, unsigned tensor_d1,
    unsigned long long stride0_elems) {
  const unsigned long long ga = (unsigned long long)(uintptr_t)gptr;
  u32x4 g0;
  g0[0] = 1u;                                          // count=1, user descriptor
  g0[1] = lds_off;                                     // lds_addr (bytes)
  g0[2] = (unsigned)(ga & 0xffffffffu);                // global_addr[31:0]
  g0[3] = (unsigned)((ga >> 32) & 0x01ffffffu) | (2u << 30);  // addr[56:32] | type=2
  i32x8 g1;
  g1[0] = (int)(1u << 16);                             // data_size=1 (2 bytes)
  g1[1] = (int)((tensor_d0 & 0xffffu) << 16);          // tensor_dim0[15:0]
  g1[2] = (int)(((tensor_d0 >> 16) & 0xffffu) | ((tensor_d1 & 0xffffu) << 16));
  g1[3] = (int)(((tensor_d1 >> 16) & 0xffffu) | ((tile_d0 & 0xffffu) << 16));
  g1[4] = (int)(tile_d1 & 0xffffu);                    // tile_dim1 (0 => 1-D), tile_dim2=0
  g1[5] = (int)(stride0_elems & 0xffffffffu);          // tensor_dim0_stride[31:0]
  g1[6] = (int)((stride0_elems >> 32) & 0xffffu);      // stride[47:32]; dim1_stride lo=0
  g1[7] = 0;
  i32x4 z4 = {};
#if __clang_major__ >= 23
  i32x8 z8 = {};
  __builtin_amdgcn_tensor_load_to_lds(g0, g1, z4, z4, z8, 0);
#else
  __builtin_amdgcn_tensor_load_to_lds(g0, g1, z4, z4, 0);
#endif
}

// ---------------- Kernel 1: per-head projection  y = x_head @ W^T ------------
__global__ __launch_bounds__(32)
void proj_kernel(const float* __restrict__ x, const float* __restrict__ W,
                 _Float16* __restrict__ out, int transposed) {
  const int tile = blockIdx.x;            // 16 flat rows of N*S
  const int h    = blockIdx.y;
  const int lane = threadIdx.x & 31;
  const int col  = lane & 15;
  const int hi   = lane >> 4;
  const int row0 = tile * 16;

  v16h a0 = frag_f32(x, EMBED, row0, h * HEAD_DIM);
  v16h a1 = frag_f32(x, EMBED, row0, h * HEAD_DIM + 32);

#pragma unroll
  for (int t = 0; t < 4; ++t) {
    v16h b0 = frag_f32(W, HEAD_DIM, t * 16, 0);   // B^T is W itself (e x d)
    v16h b1 = frag_f32(W, HEAD_DIM, t * 16, 32);
    v8f acc = {};
    acc = wmma_f16(a0, b0, acc);
    acc = wmma_f16(a1, b1, acc);
    const int d = t * 16 + col;
#pragma unroll
    for (int i = 0; i < 8; ++i) {
      const int fr = row0 + i + 8 * hi;            // flat row in [0, N*S)
      const int n  = fr >> 11;
      const int s  = fr & (SEQ - 1);
      size_t idx;
      if (transposed)
        idx = (((size_t)(n * HEADS + h)) * HEAD_DIM + d) * SEQ + s;
      else
        idx = (((size_t)(n * HEADS + h)) * SEQ + s) * HEAD_DIM + d;
      out[idx] = (_Float16)acc[i];
    }
  }
}

// ---------------- Kernel 2: flash attention with TDM-fed LDS tiles -----------
// 4 waves / block; block owns 64 query rows of one (n,h). Wave 0 drives the
// Tensor Data Mover to double-buffer 64-key K/V tiles in LDS.
__global__ __launch_bounds__(128)
void attn_kernel(const _Float16* __restrict__ Qp, const _Float16* __restrict__ Kp,
                 const _Float16* __restrict__ VpT, const int* __restrict__ mask,
                 _Float16* __restrict__ Obuf) {
  __shared__ __align__(16) _Float16 Kbuf[2][KT * HEAD_DIM];   // [key][d]
  __shared__ __align__(16) _Float16 Vbuf[2][HEAD_DIM * KT];   // [d][key]
  __shared__ __align__(16) _Float16 plds[NWAVE][16 * KT];     // per-wave P^T staging

  const int qt   = blockIdx.x;
  const int nh   = blockIdx.y;
  const int n    = nh >> 4;
  const int h    = nh & 15;
  const int tid  = threadIdx.x;
  const int wid  = tid >> 5;
  const int lane = tid & 31;
  const int col  = lane & 15;
  const int hi   = lane >> 4;
  const int q0   = qt * (16 * NWAVE) + wid * 16;

  const _Float16* Qb = Qp  + (size_t)nh * SEQ * HEAD_DIM;
  const _Float16* Kb = Kp  + (size_t)nh * SEQ * HEAD_DIM;
  const _Float16* Vb = VpT + (size_t)nh * HEAD_DIM * SEQ;
  const int*      mb = mask + (size_t)n * SEQ * SEQ;

  // Q fragments stay resident across the whole key loop.
  v16h qa0 = frag_f16(Qb, HEAD_DIM, q0, 0);
  v16h qa1 = frag_f16(Qb, HEAD_DIM, q0, 32);

  float mstat[8], lstat[8];
  v8f oacc[4];
  v8f vzero = {};
#pragma unroll
  for (int i = 0; i < 8; ++i) { mstat[i] = -__builtin_inff(); lstat[i] = 0.0f; }
#pragma unroll
  for (int t = 0; t < 4; ++t) oacc[t] = vzero;

  const float scale = 0.03125f;  // 1/sqrt(EMBED)
  const int NT = SEQ / KT;

  if (wid == 0) {   // prologue: DMA tile 0 into buffer 0
    tdm_load_f16(lds_byte_off(&Kbuf[0][0]), Kb, KT * HEAD_DIM, 0,
                 KT * HEAD_DIM, 0, KT * HEAD_DIM);                 // 1-D 8KB copy
    tdm_load_f16(lds_byte_off(&Vbuf[0][0]), Vb, KT, HEAD_DIM,
                 SEQ, HEAD_DIM, SEQ);                               // 2-D strided
  }

  for (int kt = 0; kt < NT; ++kt) {
    const int k0  = kt * KT;
    const int buf = kt & 1;
    if (wid == 0) {
      if (kt + 1 < NT) {  // DMA next tile, then wait for the current one
        tdm_load_f16(lds_byte_off(&Kbuf[buf ^ 1][0]),
                     Kb + (size_t)(k0 + KT) * HEAD_DIM,
                     KT * HEAD_DIM, 0, KT * HEAD_DIM, 0, KT * HEAD_DIM);
        tdm_load_f16(lds_byte_off(&Vbuf[buf ^ 1][0]),
                     Vb + (k0 + KT), KT, HEAD_DIM, SEQ, HEAD_DIM, SEQ);
        __builtin_amdgcn_s_wait_tensorcnt(2);   // in-order: current tile done
      } else {
        __builtin_amdgcn_s_wait_tensorcnt(0);
      }
    }
    __syncthreads();                            // tile visible to all waves
    const _Float16* Kt = &Kbuf[buf][0];
    const _Float16* Vt = &Vbuf[buf][0];
    __builtin_prefetch(mb + (size_t)q0 * SEQ + k0 + KT, 0, 1);

    // scores for 64 keys: 4 column groups of 16
    v8f s[4];
#pragma unroll
    for (int g = 0; g < 4; ++g)
      s[g] = wmma_f16(qa1, frag_f16(Kt, HEAD_DIM, g * 16, 32),
                      wmma_f16(qa0, frag_f16(Kt, HEAD_DIM, g * 16, 0), vzero));

    // mask + scale (C/D layout: lane = key column, vgpr i = row i + 8*hi)
    float e[4][8];
#pragma unroll
    for (int g = 0; g < 4; ++g)
#pragma unroll
      for (int i = 0; i < 8; ++i) {
        const int q = q0 + i + 8 * hi;
        const int m = mb[(size_t)q * SEQ + k0 + g * 16 + col];
        e[g][i] = (m == 0) ? -1e20f : s[g][i] * scale;
      }

    // online softmax: per-row max/sum across the 16 lanes sharing a row
    float mnew[8], alpha[8];
#pragma unroll
    for (int i = 0; i < 8; ++i) {
      float t = fmaxf(fmaxf(e[0][i], e[1][i]), fmaxf(e[2][i], e[3][i]));
#pragma unroll
      for (int m = 1; m < 16; m <<= 1) t = fmaxf(t, __shfl_xor(t, m, 32));
      mnew[i]  = fmaxf(mstat[i], t);
      alpha[i] = __expf(mstat[i] - mnew[i]);
      mstat[i] = mnew[i];
    }
#pragma unroll
    for (int i = 0; i < 8; ++i) {
      const int r = i + 8 * hi;
      float rs = 0.0f;
#pragma unroll
      for (int g = 0; g < 4; ++g) {
        const float p = __expf(e[g][i] - mnew[i]);
        rs += p;
        plds[wid][r * KT + g * 16 + col] = (_Float16)p;  // P^T: row=q, col=key
      }
#pragma unroll
      for (int m = 1; m < 16; m <<= 1) rs += __shfl_xor(rs, m, 32);
      lstat[i] = lstat[i] * alpha[i] + rs;
    }
    // same-wave LDS store->load ordering (private region, no barrier needed)
    asm volatile("s_wait_dscnt 0" ::: "memory");

    v16h pa0 = frag_f16(&plds[wid][0], KT, 0, 0);    // P rows x keys 0..31
    v16h pa1 = frag_f16(&plds[wid][0], KT, 0, 32);   // P rows x keys 32..63

    // O = diag(alpha)*O + P @ V   (B^T = V^T tile: rows d, cols key)
#pragma unroll
    for (int t = 0; t < 4; ++t) {
#pragma unroll
      for (int i = 0; i < 8; ++i) oacc[t][i] *= alpha[i];
      oacc[t] = wmma_f16(pa0, frag_f16(Vt, KT, t * 16, 0), oacc[t]);
      oacc[t] = wmma_f16(pa1, frag_f16(Vt, KT, t * 16, 32), oacc[t]);
    }
    __syncthreads();   // all waves done with this buffer before DMA reuses it
  }

  // normalize and store concatenated heads: Obuf[n][s][h*64 + d] (f16)
#pragma unroll
  for (int t = 0; t < 4; ++t) {
    const int d = h * HEAD_DIM + t * 16 + col;
#pragma unroll
    for (int i = 0; i < 8; ++i) {
      const int s = q0 + i + 8 * hi;
      const float o = oacc[t][i] / lstat[i];
      Obuf[((size_t)n * SEQ + s) * EMBED + d] = (_Float16)o;
    }
  }
}

// ---------------- Kernel 3: output projection  out = O @ Wo^T + bo ----------
__global__ __launch_bounds__(32)
void ogemm_kernel(const _Float16* __restrict__ Obuf, const float* __restrict__ Wo,
                  const float* __restrict__ bo, float* __restrict__ out) {
  const int mt   = blockIdx.x;       // 16-row tile over N*S
  const int et   = blockIdx.y;       // 64-wide output tile
  const int lane = threadIdx.x & 31;
  const int col  = lane & 15;
  const int hi   = lane >> 4;
  const int m0   = mt * 16;
  const int eb   = et * 64;

  v8f acc[4];
  v8f vzero = {};
#pragma unroll
  for (int t = 0; t < 4; ++t) acc[t] = vzero;

  for (int kc = 0; kc < EMBED; kc += 32) {
    v16h a = frag_f16(Obuf, EMBED, m0, kc);
#pragma unroll
    for (int t = 0; t < 4; ++t) {
      v16h b = frag_f32(Wo, EMBED, eb + t * 16, kc);   // B^T = Wo (e_out x e_in)
      acc[t] = wmma_f16(a, b, acc[t]);
    }
  }

#pragma unroll
  for (int t = 0; t < 4; ++t) {
    const int e = eb + t * 16 + col;
    const float bias = bo[e];
#pragma unroll
    for (int i = 0; i < 8; ++i) {
      const int r = m0 + i + 8 * hi;
      out[(size_t)r * EMBED + e] = acc[t][i] + bias;
    }
  }
}

extern "C" void kernel_launch(void* const* d_in, const int* in_sizes, int n_in,
                              void* d_out, int out_size, void* d_ws, size_t ws_size,
                              hipStream_t stream) {
  const float* values_x  = (const float*)d_in[0];
  const float* keys_x    = (const float*)d_in[1];
  const float* queries_x = (const float*)d_in[2];
  const int*   mask      = (const int*)d_in[3];
  const float* Wv        = (const float*)d_in[4];
  const float* Wk        = (const float*)d_in[5];
  const float* Wq        = (const float*)d_in[6];
  const float* Wo        = (const float*)d_in[7];
  const float* bo        = (const float*)d_in[8];
  float*       out       = (float*)d_out;

  const size_t NHSD = (size_t)NB * HEADS * SEQ * HEAD_DIM;   // 8.4M elems
  _Float16* Qp   = (_Float16*)d_ws;       // [n,h,s,d] f16  (16 MB)
  _Float16* Kp   = Qp  + NHSD;            // [n,h,s,d] f16  (16 MB)
  _Float16* VpT  = Kp  + NHSD;            // [n,h,d,s] f16  (16 MB)
  _Float16* Obuf = VpT + NHSD;            // [n,s,e]   f16  (16 MB)

  dim3 gp(NB * SEQ / 16, HEADS);
  proj_kernel<<<gp, 32, 0, stream>>>(queries_x, Wq, Qp, 0);
  proj_kernel<<<gp, 32, 0, stream>>>(keys_x,    Wk, Kp, 0);
  proj_kernel<<<gp, 32, 0, stream>>>(values_x,  Wv, VpT, 1);

  attn_kernel<<<dim3(SEQ / (16 * NWAVE), NB * HEADS), 128, 0, stream>>>(
      Qp, Kp, VpT, mask, Obuf);

  ogemm_kernel<<<dim3(NB * SEQ / 16, EMBED / 64), 32, 0, stream>>>(Obuf, Wo, bo, out);
}
